// VectorQuantizer_47682726920786
// MI455X (gfx1250) — compile-verified, matrix operand order checked by reading, not run on hardware
//
#include <hip/hip_runtime.h>
#include <hip/hip_bf16.h>

// B=64, T=1024, D=64, K=512
#define BT      65536
#define DIM     64
#define KCB     512
#define ROWS_PER_WG 128   // 8 waves * 16 rows
#define THREADS 256

typedef __attribute__((ext_vector_type(16))) __bf16 v16bf;
typedef __attribute__((ext_vector_type(8)))  float  v8f;

struct Bf16Pair { __bf16 h; __bf16 l; };

// Split x into bf16 hi + lo (x ~= hi + lo). The A side is negated before the
// split so WMMA accumulates 0.5*e_sq - cross directly (bf16 WMMA has no A/B neg).
static __device__ __forceinline__ Bf16Pair bf16_split(float x) {
    Bf16Pair p;
    __bf16 hh = (__bf16)x;
    p.h = hh;
    p.l = (__bf16)(x - (float)hh);
    return p;
}

// Permute K-group order so one contiguous 32B LDS read yields the v16bf
// B-operand for a given (k-chunk, lane-half):
// lds group order per 32-K chunk: [0,2,1,3] i.e. K 0-7,16-23 then 8-15,24-31.
static __device__ __forceinline__ int lds_group(int g) {
    return (g & 4) | ((g & 1) << 1) | ((g & 2) >> 1);
}

__global__ __launch_bounds__(THREADS)
void vq_wmma_kernel(const float* __restrict__ codes,
                    const float* __restrict__ codebook,
                    float* __restrict__ out) {
    // LDS: codebook bf16 hi/lo planes in WMMA-B layout + 0.5*||e||^2 + winners
    __shared__ __align__(64) __bf16 sHi[KCB * DIM];   // 64 KB
    __shared__ __align__(64) __bf16 sLo[KCB * DIM];   // 64 KB
    __shared__ float sESqHalf[KCB];                   // 2 KB
    __shared__ int   sIdx[ROWS_PER_WG];               // 0.5 KB

    const int tid = threadIdx.x;

    // ---- Phase 1: convert codebook f32 -> bf16 hi/lo into LDS (permuted) ----
    // 512 rows / 256 threads = 2 rows each.
    for (int r = tid; r < KCB; r += THREADS) {
        const float* src = codebook + r * DIM;
        float sumsq = 0.0f;
        #pragma unroll
        for (int d = 0; d < DIM; ++d) {
            float x = src[d];
            sumsq = fmaf(x, x, sumsq);
            Bf16Pair p = bf16_split(x);
            int g   = d >> 3;
            int pos = r * DIM + lds_group(g) * 8 + (d & 7);
            sHi[pos] = p.h;
            sLo[pos] = p.l;
        }
        sESqHalf[r] = 0.5f * sumsq;
    }
    __syncthreads();

    // ---- Phase 2: per-wave 16-row tile, sweep all 512 codebook columns ----
    const int lane   = tid & 31;
    const int waveId = tid >> 5;
    const int n16    = lane & 15;      // column within 16-wide tile / row id
    const int half   = lane >> 4;      // lane-half selects K sub-block
    const int rowBase = blockIdx.x * ROWS_PER_WG + waveId * 16;
    const float* crow = codes + (rowBase + n16) * DIM;

    // Build A operands (negated codes, hi/lo split) for both 32-K chunks.
    // 16-bit A layout: lanes 0-15 -> elems[0..7]=K(kb..kb+7), [8..15]=K(kb+16..kb+23)
    // with kb = kchunk*32 (+8 for lanes 16-31).
    v16bf aHi[2], aLo[2];
    #pragma unroll
    for (int kc = 0; kc < 2; ++kc) {
        const int kb = kc * 32 + half * 8;
        #pragma unroll
        for (int e = 0; e < 8; ++e) {
            Bf16Pair p0 = bf16_split(-crow[kb + e]);
            Bf16Pair p1 = bf16_split(-crow[kb + 16 + e]);
            aHi[kc][e]     = p0.h;  aLo[kc][e]     = p0.l;
            aHi[kc][e + 8] = p1.h;  aLo[kc][e + 8] = p1.l;
        }
    }

    float bestV[8];
    int   bestI[8];
    #pragma unroll
    for (int r = 0; r < 8; ++r) { bestV[r] = 3.4e38f; bestI[r] = 0; }

    // Unroll by 2: two independent accumulator chains + LDS load streams let
    // the scheduler fill the WMMA->VALU hazard slots (4 co-exec nops) and
    // overlap ds_load latency with the other tile's WMMAs.
    #pragma unroll 2
    for (int ct = 0; ct < KCB / 16; ++ct) {
        const int n = ct * 16 + n16;                 // this lane's column
        const float h = sESqHalf[n];
        v8f c = {h, h, h, h, h, h, h, h};            // init C with 0.5*||e||^2

        #pragma unroll
        for (int kc = 0; kc < 2; ++kc) {
            const int off = n * DIM + (kc * 2 + half) * 16;
            const v16bf bh = *(const v16bf*)(sHi + off);
            const v16bf bl = *(const v16bf*)(sLo + off);
            // score += (-codes_hi)*cb_hi + (-codes_hi)*cb_lo + (-codes_lo)*cb_hi
            c = __builtin_amdgcn_wmma_f32_16x16x32_bf16(false, aHi[kc], false, bh,
                                                        (short)0, c, false, false);
            c = __builtin_amdgcn_wmma_f32_16x16x32_bf16(false, aHi[kc], false, bl,
                                                        (short)0, c, false, false);
            c = __builtin_amdgcn_wmma_f32_16x16x32_bf16(false, aLo[kc], false, bh,
                                                        (short)0, c, false, false);
        }

        // Running argmin; strict < keeps the earliest (lowest) index per lane.
        #pragma unroll
        for (int r = 0; r < 8; ++r) {
            if (c[r] < bestV[r]) { bestV[r] = c[r]; bestI[r] = n; }
        }
    }

    // ---- Phase 3: cross-lane lexicographic min within each 16-lane half ----
    // C layout: vgpr r <-> row r (lanes 0-15) / row r+8 (lanes 16-31).
    #pragma unroll
    for (int off = 8; off >= 1; off >>= 1) {
        #pragma unroll
        for (int r = 0; r < 8; ++r) {
            float ov = __shfl_xor(bestV[r], off, 32);
            int   oi = __shfl_xor(bestI[r], off, 32);
            if (ov < bestV[r] || (ov == bestV[r] && oi < bestI[r])) {
                bestV[r] = ov; bestI[r] = oi;
            }
        }
    }
    if (n16 == 0) {
        const int base = waveId * 16 + half * 8;
        #pragma unroll
        for (int r = 0; r < 8; ++r) sIdx[base + r] = bestI[r];
    }
    __syncthreads();

    // ---- Phase 4: gather winning codebook rows (L2-hot), coalesced write ----
    const float2* cbv  = (const float2*)codebook;
    float2*       outv = (float2*)out;
    #pragma unroll 4
    for (int m = 0; m < 16; ++m) {
        const int idx = sIdx[waveId * 16 + m];
        const int g   = rowBase + m;
        outv[g * (DIM / 2) + lane] = cbv[idx * (DIM / 2) + lane];
    }
}

extern "C" void kernel_launch(void* const* d_in, const int* in_sizes, int n_in,
                              void* d_out, int out_size, void* d_ws, size_t ws_size,
                              hipStream_t stream) {
    (void)in_sizes; (void)n_in; (void)d_ws; (void)ws_size; (void)out_size;
    const float* codes    = (const float*)d_in[0];
    const float* codebook = (const float*)d_in[1];
    float*       out      = (float*)d_out;
    dim3 grid(BT / ROWS_PER_WG);   // 512
    dim3 block(THREADS);           // 256 = 8 waves
    vq_wmma_kernel<<<grid, block, 0, stream>>>(codes, codebook, out);
}
